// GraphAttentionLayer_28544352649382
// MI455X (gfx1250) — matrix operand order repaired and checked by reading, not compiled, not run on hardware
//
#include <hip/hip_runtime.h>
#include <hip/hip_bf16.h>
#include <math.h>

typedef __attribute__((ext_vector_type(16))) _Float16 v16h;
typedef __attribute__((ext_vector_type(8)))  float    v8f;

#define N_NODES 8192
#define IN_F    128
#define OUT_F   64
#define LRELU_ALPHA 0.2f

// ---------------- workspace layout ----------------
// [0            , 1 MB)   : hT  (f16)  [OUT_F][N_NODES]   (transposed h, WMMA B source)
// [1 MB         , +32 KB) : h_i (f32)  [N_NODES]
// [+32 KB       , +32 KB) : h_j (f32)  [N_NODES]
// [+4 B]                  : Menc (ordered-int encoded global max of h_j)

__device__ __forceinline__ int enc_ord(float f) {
    int i = __float_as_int(f);
    return (i >= 0) ? i : (i ^ 0x7fffffff);
}

// -------------------------------------------------------------------------
// Kernel 1: h = x @ W (W staged in LDS), write hT in f16, reduce h_i / h_j
// per row, and track a global stability bound max_j h_j via atomic ordered max.
// -------------------------------------------------------------------------
__global__ __launch_bounds__(256) void gat_h_kernel(
    const float* __restrict__ x, const float* __restrict__ W,
    const float* __restrict__ a,
    _Float16* __restrict__ hT, float* __restrict__ hi, float* __restrict__ hj,
    int* __restrict__ Menc)
{
    __shared__ float sW[IN_F * OUT_F];   // 32 KB
    __shared__ float s1[256];
    __shared__ float s2[256];
    __shared__ float bmax[4];

    const int tid = threadIdx.x;
    for (int t = tid; t < IN_F * OUT_F; t += 256) sW[t] = W[t];
    __syncthreads();

    const int o  = tid & 63;      // output feature
    const int ii = tid >> 6;      // row within block
    const int i  = blockIdx.x * 4 + ii;

    const float* xr = x + (size_t)i * IN_F;
    float acc = 0.f;
    #pragma unroll 8
    for (int k = 0; k < IN_F; ++k)
        acc = fmaf(xr[k], sW[k * OUT_F + o], acc);

    hT[(size_t)o * N_NODES + i] = (_Float16)acc;

    s1[tid] = acc * a[o];
    s2[tid] = acc * a[OUT_F + o];
    __syncthreads();

    if (o == 0) {
        float r1 = 0.f, r2 = 0.f;
        #pragma unroll 8
        for (int t = 0; t < 64; ++t) { r1 += s1[ii * 64 + t]; r2 += s2[ii * 64 + t]; }
        hi[i] = r1;
        hj[i] = r2;
        bmax[ii] = r2;
    }
    __syncthreads();
    if (tid == 0) {
        float m = fmaxf(fmaxf(bmax[0], bmax[1]), fmaxf(bmax[2], bmax[3]));
        atomicMax(Menc, enc_ord(m));
    }
}

// -------------------------------------------------------------------------
// Software-pipelined tile of all per-iteration global inputs.
// -------------------------------------------------------------------------
struct Tile {
    float4 a0, a1, a2, a3;    // adjacency slots (A-matrix pattern)
    float4 j0, j1, j2, j3;    // h_j slots (same pattern)
    float4 b[8];              // four 32x16 f16 B tiles, 2 x 16B each
};

__device__ __forceinline__ void load_tile(
    Tile& T, const float* __restrict__ arow, const float* __restrict__ hj,
    const _Float16* __restrict__ hb0, const _Float16* __restrict__ hb1,
    const _Float16* __restrict__ hb2, const _Float16* __restrict__ hb3,
    int jb, int k0)
{
    T.a0 = *(const float4*)(arow + jb + k0);
    T.a1 = *(const float4*)(arow + jb + k0 + 4);
    T.a2 = *(const float4*)(arow + jb + k0 + 16);
    T.a3 = *(const float4*)(arow + jb + k0 + 20);
    T.j0 = *(const float4*)(hj + jb + k0);
    T.j1 = *(const float4*)(hj + jb + k0 + 4);
    T.j2 = *(const float4*)(hj + jb + k0 + 16);
    T.j3 = *(const float4*)(hj + jb + k0 + 20);
    T.b[0] = *(const float4*)(hb0 + jb); T.b[1] = *(const float4*)(hb0 + jb + 16);
    T.b[2] = *(const float4*)(hb1 + jb); T.b[3] = *(const float4*)(hb1 + jb + 16);
    T.b[4] = *(const float4*)(hb2 + jb); T.b[5] = *(const float4*)(hb2 + jb + 16);
    T.b[6] = *(const float4*)(hb3 + jb); T.b[7] = *(const float4*)(hb3 + jb + 16);
}

// -------------------------------------------------------------------------
// Kernel 2: fused masked-softmax attention aggregation (flash-GAT).
// Block = 128 threads (4 waves), one 16-row tile per block.
// Wave w sweeps columns [w*2048, (w+1)*2048) in K=32 tiles, with the next
// tile's 16 loads in flight while the current tile's VALU + 4 WMMAs execute.
// Fixed global stability bound M => partials across waves simply add.
// -------------------------------------------------------------------------
__global__ __launch_bounds__(128) void gat_attn_kernel(
    const float* __restrict__ adj,
    const _Float16* __restrict__ hT,
    const float* __restrict__ hi, const float* __restrict__ hj,
    const int* __restrict__ Menc,
    float* __restrict__ out)
{
    __shared__ float accbuf[4 * 16 * 64];  // 16 KB : per-wave partial numerators
    __shared__ float sumbuf[4 * 16];       // per-wave partial row sums

    const int tid  = threadIdx.x;
    const int w    = tid >> 5;     // wave in block
    const int lane = tid & 31;
    const int m    = lane & 15;    // A row / B column within tile
    const int hs   = lane >> 4;    // lane half
    const int k0   = hs * 8;       // A-matrix K offset pattern (ISA 7.12.2)
    const int klo  = hs * 8;       // B-matrix K offset pattern (ISA 7.12.4)
    const int rowBase = blockIdx.x * 16;

    int eM = Menc[0];
    if (eM < 0) eM ^= 0x7fffffff;
    const float Mval = __int_as_float(eM);              // global bound on h_j
    const float si   = hi[rowBase + m];                 // per-row a1.h_i

    const float*    arow = adj + (size_t)(rowBase + m) * N_NODES;
    const _Float16* hb0  = hT + (size_t)(m)      * N_NODES + klo;
    const _Float16* hb1  = hT + (size_t)(16 + m) * N_NODES + klo;
    const _Float16* hb2  = hT + (size_t)(32 + m) * N_NODES + klo;
    const _Float16* hb3  = hT + (size_t)(48 + m) * N_NODES + klo;

    v8f acc0 = {}, acc1 = {}, acc2 = {}, acc3 = {};
    float rs0 = 0.f, rs1 = 0.f, rs2 = 0.f, rs3 = 0.f;  // independent sum chains

    const int JT0 = w * (N_NODES / 4);
    const int JT1 = JT0 + (N_NODES / 4);

    Tile cur, nxt;
    load_tile(cur, arow, hj, hb0, hb1, hb2, hb3, JT0, k0);

    #pragma unroll 2
    for (int jb = JT0; jb < JT1; jb += 32) {
        // issue ALL of next tile's loads before touching this tile's data
        const int jn = (jb + 32 < JT1) ? (jb + 32) : JT0;   // wrap keeps addr valid
        load_tile(nxt, arow, hj, hb0, hb1, hb2, hb3, jn, k0);
        __builtin_prefetch(arow + jb + 96 + k0, 0, 3);      // global_prefetch_b8

        const float av[16] = {cur.a0.x, cur.a0.y, cur.a0.z, cur.a0.w,
                              cur.a1.x, cur.a1.y, cur.a1.z, cur.a1.w,
                              cur.a2.x, cur.a2.y, cur.a2.z, cur.a2.w,
                              cur.a3.x, cur.a3.y, cur.a3.z, cur.a3.w};
        const float jv[16] = {cur.j0.x, cur.j0.y, cur.j0.z, cur.j0.w,
                              cur.j1.x, cur.j1.y, cur.j1.z, cur.j1.w,
                              cur.j2.x, cur.j2.y, cur.j2.z, cur.j2.w,
                              cur.j3.x, cur.j3.y, cur.j3.z, cur.j3.w};

        float pvv[16];
        #pragma unroll
        for (int t = 0; t < 16; ++t) {
            float ev  = si + jv[t];
            ev        = fmaxf(ev, LRELU_ALPHA * ev);     // branchless leaky relu
            float msk = (av[t] > 0.f) ? 1.f : 0.f;       // v_cndmask, inline consts
            pvv[t]    = __expf(ev - Mval) * msk;         // exp always computed
        }

        union { v16h v; _Float16 h[16]; } A;
        #pragma unroll
        for (int t = 0; t < 16; t += 4) {                // 4 independent sum chains
            rs0 += pvv[t + 0];
            rs1 += pvv[t + 1];
            rs2 += pvv[t + 2];
            rs3 += pvv[t + 3];
        }
        #pragma unroll
        for (int t = 0; t < 16; ++t) A.h[t] = (_Float16)pvv[t];

        union BU { float4 f[2]; v16h v; } B0, B1, B2, B3;
        B0.f[0] = cur.b[0]; B0.f[1] = cur.b[1];
        B1.f[0] = cur.b[2]; B1.f[1] = cur.b[3];
        B2.f[0] = cur.b[4]; B2.f[1] = cur.b[5];
        B3.f[0] = cur.b[6]; B3.f[1] = cur.b[7];

        acc0 = __builtin_amdgcn_wmma_f32_16x16x32_f16(false, A.v, false, B0.v,
                                                      (short)0, acc0, false, false);
        acc1 = __builtin_amdgcn_wmma_f32_16x16x32_f16(false, A.v, false, B1.v,
                                                      (short)0, acc1, false, false);
        acc2 = __builtin_amdgcn_wmma_f32_16x16x32_f16(false, A.v, false, B2.v,
                                                      (short)0, acc2, false, false);
        acc3 = __builtin_amdgcn_wmma_f32_16x16x32_f16(false, A.v, false, B3.v,
                                                      (short)0, acc3, false, false);

        cur = nxt;   // register rotation; folds away under unroll-by-2
    }

    float rowsum = (rs0 + rs1) + (rs2 + rs3);
    // combine the two lane-halves of each row's sum (disjoint K coverage)
    rowsum += __shfl_xor(rowsum, 16, 32);
    if (hs == 0) sumbuf[w * 16 + m] = rowsum;

    // spill per-wave partial numerators (C layout: VGPR v -> row v + hs*8)
    #pragma unroll
    for (int vv = 0; vv < 8; ++vv) {
        const int r = vv + hs * 8;
        accbuf[(w * 16 + r) * 64 +  0 + m] = acc0[vv];
        accbuf[(w * 16 + r) * 64 + 16 + m] = acc1[vv];
        accbuf[(w * 16 + r) * 64 + 32 + m] = acc2[vv];
        accbuf[(w * 16 + r) * 64 + 48 + m] = acc3[vv];
    }
    __syncthreads();

    // cross-wave reduction + softmax normalization + ELU
    for (int idx = tid; idx < 16 * 64; idx += 128) {
        const int r = idx >> 6;
        const int c = idx & 63;
        float num = accbuf[r * 64 + c] + accbuf[(16 + r) * 64 + c]
                  + accbuf[(32 + r) * 64 + c] + accbuf[(48 + r) * 64 + c];
        float den = sumbuf[r] + sumbuf[16 + r] + sumbuf[32 + r] + sumbuf[48 + r];
        float oV = num / den;
        oV = (oV > 0.f) ? oV : (__expf(oV) - 1.f);   // ELU
        out[(size_t)(rowBase + r) * OUT_F + c] = oV;
    }
}

// -------------------------------------------------------------------------
extern "C" void kernel_launch(void* const* d_in, const int* in_sizes, int n_in,
                              void* d_out, int out_size, void* d_ws, size_t ws_size,
                              hipStream_t stream)
{
    const float* x   = (const float*)d_in[0];
    const float* adj = (const float*)d_in[1];
    const float* W   = (const float*)d_in[2];
    const float* a   = (const float*)d_in[3];
    float* out = (float*)d_out;

    char* ws = (char*)d_ws;
    _Float16* hT = (_Float16*)ws;                                   // 1 MB
    float*    hi = (float*)(ws + (size_t)OUT_F * N_NODES * sizeof(_Float16));
    float*    hj = hi + N_NODES;
    int*    Menc = (int*)(hj + N_NODES);

    hipMemsetAsync(Menc, 0, sizeof(int), stream);

    gat_h_kernel<<<N_NODES / 4, 256, 0, stream>>>(x, W, a, hT, hi, hj, Menc);
    gat_attn_kernel<<<N_NODES / 16, 128, 0, stream>>>(adj, hT, hi, hj, Menc, out);
}
